// MultiHeadSelfAttention_46909632807312
// MI455X (gfx1250) — matrix operand order repaired and checked by reading, not compiled, not run on hardware
//
#include <hip/hip_runtime.h>

// ---------------------------------------------------------------------------
// MI455X (gfx1250) attention block: GroupNorm -> QKV GEMM -> MHSA -> OutProj
// Matmuls via v_wmma_f32_16x16x32_f16 (wave32, f32 accumulate); contiguous
// global->LDS staging via global_load_async_to_lds_b128 (ASYNCcnt).
// ---------------------------------------------------------------------------

typedef _Float16 f16;
typedef _Float16 v16h __attribute__((ext_vector_type(16)));
typedef _Float16 v8h  __attribute__((ext_vector_type(8)));
typedef float    v8f  __attribute__((ext_vector_type(8)));

#define B_  4
#define S_  2048
#define C_  1024
#define H_  16
#define D_  64
#define NG_ 32

__device__ inline v16h join8(v8h lo, v8h hi) {
  v16h r;
#pragma unroll
  for (int i = 0; i < 8; ++i) { r[i] = lo[i]; r[i + 8] = hi[i]; }
  return r;
}

__device__ inline v8f wmma16x16x32(v16h a, v16h b, v8f c) {
  // (neg_a, A, neg_b, B, c_mod, C, reuse_a, reuse_b)
  return __builtin_amdgcn_wmma_f32_16x16x32_f16(false, a, false, b,
                                                (short)0, c, false, false);
}

// Async global->LDS copy, 16 bytes per lane. VDST = per-lane LDS byte address
// (generic shared pointers carry the LDS offset in their low 32 bits per the
// gfx1250 aperture mapping); VADDR = 64-bit global address. Tracked by
// ASYNCcnt; pair with s_wait_asynccnt before the consuming barrier.
__device__ inline void async_ld_b128(void* lds, const void* g) {
  asm volatile("global_load_async_to_lds_b128 %0, %1, off"
               :: "v"((unsigned)(size_t)lds), "v"(g)
               : "memory");
}
__device__ inline void async_wait0() {
  asm volatile("s_wait_asynccnt 0x0" ::: "memory");
}

// ---------------------------------------------------------------------------
// 1) GroupNorm statistics: one block per (batch, group); reduce S*32 elems.
// ---------------------------------------------------------------------------
__global__ __launch_bounds__(256)
void gn_stats_kernel(const float* __restrict__ x, float* __restrict__ stats) {
  const int bg = blockIdx.x;            // 0..127
  const int b = bg >> 5, g = bg & 31;
  const float* base = x + (size_t)b * S_ * C_ + g * 32;
  float s = 0.f, ss = 0.f;
  for (int i = threadIdx.x; i < S_ * 32; i += 256) {
    const int srow = i >> 5, c = i & 31;
    const float v = base[(size_t)srow * C_ + c];
    s += v; ss += v * v;
  }
  __shared__ float red0[256], red1[256];
  red0[threadIdx.x] = s; red1[threadIdx.x] = ss;
  __syncthreads();
  for (int st = 128; st > 0; st >>= 1) {
    if (threadIdx.x < st) {
      red0[threadIdx.x] += red0[threadIdx.x + st];
      red1[threadIdx.x] += red1[threadIdx.x + st];
    }
    __syncthreads();
  }
  if (threadIdx.x == 0) {
    const float n = (float)(S_ * 32);
    const float mean = red0[0] / n;
    const float var  = red1[0] / n - mean * mean;
    stats[bg * 2 + 0] = mean;
    stats[bg * 2 + 1] = rsqrtf(var + 1e-6f);
  }
}

// ---------------------------------------------------------------------------
// 2) GroupNorm apply + convert to f16 (4 elems/thread).
// ---------------------------------------------------------------------------
__global__ __launch_bounds__(256)
void gn_apply_kernel(const float* __restrict__ x, const float* __restrict__ gamma,
                     const float* __restrict__ beta, const float* __restrict__ stats,
                     f16* __restrict__ h) {
  const size_t i = ((size_t)blockIdx.x * 256 + threadIdx.x) * 4;
  const int c = (int)(i % C_);
  const size_t row = i / C_;            // b*S + s
  const int b = (int)(row / S_);
  const int g = c >> 5;                 // 4 consecutive c share one group
  const float mean = stats[(b * 32 + g) * 2 + 0];
  const float rstd = stats[(b * 32 + g) * 2 + 1];
  const float4 xv = *(const float4*)(x + i);
  h[i + 0] = (f16)((xv.x - mean) * rstd * gamma[c + 0] + beta[c + 0]);
  h[i + 1] = (f16)((xv.y - mean) * rstd * gamma[c + 1] + beta[c + 1]);
  h[i + 2] = (f16)((xv.z - mean) * rstd * gamma[c + 2] + beta[c + 2]);
  h[i + 3] = (f16)((xv.w - mean) * rstd * gamma[c + 3] + beta[c + 3]);
}

// ---------------------------------------------------------------------------
// 3) f32 -> f16 weight conversion.
// ---------------------------------------------------------------------------
__global__ __launch_bounds__(256)
void cvt_f16_kernel(const float* __restrict__ in, f16* __restrict__ out, int n) {
  const int i = (blockIdx.x * 256 + threadIdx.x) * 4;
  if (i >= n) return;
  const float4 v = *(const float4*)(in + i);
  out[i + 0] = (f16)v.x; out[i + 1] = (f16)v.y;
  out[i + 2] = (f16)v.z; out[i + 3] = (f16)v.w;
}

// ---------------------------------------------------------------------------
// 4/6) Tiled WMMA GEMM: C[M,N] = A[M,K] f16 row-major * B[K,N] f16 row-major
// + bias (+ residual). Block = 128 thr = 4 waves; block tile 128x64.
// Each wave owns 32 rows x 64 cols = 8 accum tiles; K-step 32; A staged via
// async LDS loads, B staged transposed so every fragment is two 16B ds reads.
// ---------------------------------------------------------------------------
__global__ __launch_bounds__(128)
void gemm_f16_kernel(const f16* __restrict__ A, const f16* __restrict__ Bm,
                     const float* __restrict__ bias, const float* __restrict__ residual,
                     void* __restrict__ Out, int M, int N, int K, int out_f32) {
  __shared__ __align__(16) f16 As[128][32];  // row-major [m][k], 8 KB
  __shared__ __align__(16) f16 Bs[64][32];   // transposed [n][k], 4 KB
  const int t = threadIdx.x;
  const int lane = t & 31, wave = t >> 5;
  const int hi = lane >> 4, lo = lane & 15;
  const int n0 = blockIdx.x * 64, m0 = blockIdx.y * 128;

  v8f acc[2][4] = {};
  for (int k0 = 0; k0 < K; k0 += 32) {
    __syncthreads();
    {  // stage A 128x32: thread t owns row t, 4 async 16B transfers
      const f16* src = A + (size_t)(m0 + t) * K + k0;
#pragma unroll
      for (int i = 0; i < 4; ++i)
        async_ld_b128(&As[t][i * 8], src + i * 8);
    }
    {  // stage B 32x64 transposed: thread -> (k = t/4, 16 cols at (t&3)*16)
      const int kk = t >> 2, nn = (t & 3) * 16;
      const v8h* src = (const v8h*)(Bm + (size_t)(k0 + kk) * N + n0 + nn);
      const v8h b0 = src[0], b1 = src[1];
#pragma unroll
      for (int i = 0; i < 8; ++i) {
        Bs[nn + i][kk]     = b0[i];
        Bs[nn + 8 + i][kk] = b1[i];
      }
    }
    async_wait0();
    __syncthreads();
    // A fragments (16x32): row = lane%16; K lo/hi halves split by lane/16.
    v16h af[2];
#pragma unroll
    for (int i = 0; i < 2; ++i) {
      const int arow = wave * 32 + i * 16 + lo;
      af[i] = join8(*(const v8h*)&As[arow][hi * 8],
                    *(const v8h*)&As[arow][16 + hi * 8]);
    }
#pragma unroll
    for (int j = 0; j < 4; ++j) {
      // B fragment (32x16): N = lane%16; K lo/hi halves split by lane/16.
      const int brow = j * 16 + lo;
      const v16h bf = join8(*(const v8h*)&Bs[brow][hi * 8],
                            *(const v8h*)&Bs[brow][16 + hi * 8]);
#pragma unroll
      for (int i = 0; i < 2; ++i)
        acc[i][j] = wmma16x16x32(af[i], bf, acc[i][j]);
    }
  }
  // Epilogue. C layout: N = lane%16, M = r + 8*(lane/16).
#pragma unroll
  for (int i = 0; i < 2; ++i) {
#pragma unroll
    for (int j = 0; j < 4; ++j) {
#pragma unroll
      for (int r = 0; r < 8; ++r) {
        const int row = m0 + wave * 32 + i * 16 + r + hi * 8;
        const int col = n0 + j * 16 + lo;
        const float v = acc[i][j][r] + bias[col];
        if (out_f32) {
          ((float*)Out)[(size_t)row * N + col] = v + residual[(size_t)row * N + col];
        } else {
          ((f16*)Out)[(size_t)row * N + col] = (f16)v;
        }
      }
    }
  }
}

// ---------------------------------------------------------------------------
// 5) Flash attention. qkv: f16 [B*S, 3C]; q at col h*64, k at 1024+h*64,
// v at 2048+h*64. Block = 4 waves = 64 queries per (qblock, head, batch).
// Keys processed 32 at a time; online softmax in f32. Q/K staged via async
// LDS loads; V staged transposed through registers.
// ---------------------------------------------------------------------------
__global__ __launch_bounds__(128)
void attention_kernel(const f16* __restrict__ qkv, f16* __restrict__ attn_out) {
  const int qb = blockIdx.x, h = blockIdx.y, b = blockIdx.z;
  const int t = threadIdx.x, lane = t & 31, wave = t >> 5;
  const int hi = lane >> 4, lo = lane & 15;

  __shared__ __align__(16) f16 Qs[64][64];      // [q][d]
  __shared__ __align__(16) f16 Ks[32][64];      // [key][d]
  __shared__ __align__(16) f16 Vt[64][32];      // [d][key]
  __shared__ __align__(16) f16 Ps[4][16][32];   // per-wave probs [q][key]

  const size_t rs = 3 * C_;  // qkv row stride in halfs
  const f16* qbase = qkv + ((size_t)b * S_ + qb * 64) * rs + h * D_;
  const f16* kbase = qkv + (size_t)b * S_ * rs + C_ + h * D_;
  const f16* vbase = kbase + C_;

  {  // async-load Q tile 64x64 once: thread -> (row = t/2, 32 halfs)
    const int row = t >> 1, c0 = (t & 1) * 32;
    const f16* src = qbase + (size_t)row * rs + c0;
    async_ld_b128(&Qs[row][c0 + 0], src + 0);
    async_ld_b128(&Qs[row][c0 + 8], src + 8);
    async_ld_b128(&Qs[row][c0 + 16], src + 16);
    async_ld_b128(&Qs[row][c0 + 24], src + 24);
  }

  float m[8], l[8];
  v8f O[4] = {};
#pragma unroll
  for (int r = 0; r < 8; ++r) { m[r] = -1e30f; l[r] = 0.f; }

  for (int kt = 0; kt < S_; kt += 32) {
    __syncthreads();  // protect K/V/P from previous iteration's readers
    {  // stage K (async, row-major) and V (transposed): (kr = t/4, 16 d each)
      const int kr = t >> 2, c0 = (t & 3) * 16;
      const f16* ksrc = kbase + (size_t)(kt + kr) * rs + c0;
      async_ld_b128(&Ks[kr][c0 + 0], ksrc + 0);
      async_ld_b128(&Ks[kr][c0 + 8], ksrc + 8);
      const v8h* vsrc = (const v8h*)(vbase + (size_t)(kt + kr) * rs + c0);
      const v8h v0 = vsrc[0], v1 = vsrc[1];
#pragma unroll
      for (int i = 0; i < 8; ++i) {
        Vt[c0 + i][kr]     = v0[i];
        Vt[c0 + 8 + i][kr] = v1[i];
      }
    }
    async_wait0();    // covers this K tile and (first pass) the Q tile
    __syncthreads();

    // Scores: two 16(q)x16(key) tiles, accumulated over d = 64 (2 K-steps).
    v8f s0 = {}, s1 = {};
    const int qrow = wave * 16 + lo;
#pragma unroll
    for (int d0 = 0; d0 < 64; d0 += 32) {
      const v16h aq = join8(*(const v8h*)&Qs[qrow][d0 + hi * 8],
                            *(const v8h*)&Qs[qrow][d0 + 16 + hi * 8]);
      const v16h bk0 = join8(*(const v8h*)&Ks[lo][d0 + hi * 8],
                             *(const v8h*)&Ks[lo][d0 + 16 + hi * 8]);
      const v16h bk1 = join8(*(const v8h*)&Ks[16 + lo][d0 + hi * 8],
                             *(const v8h*)&Ks[16 + lo][d0 + 16 + hi * 8]);
      s0 = wmma16x16x32(aq, bk0, s0);
      s1 = wmma16x16x32(aq, bk1, s1);
    }

    // Online softmax. Row M = r + 8*hi lives across 16 lanes of this half.
#pragma unroll
    for (int r = 0; r < 8; ++r) {
      const float x0 = s0[r] * 0.125f;   // 1/sqrt(64)
      const float x1 = s1[r] * 0.125f;
      float mx = fmaxf(x0, x1);
      mx = fmaxf(mx, __shfl_xor(mx, 1));
      mx = fmaxf(mx, __shfl_xor(mx, 2));
      mx = fmaxf(mx, __shfl_xor(mx, 4));
      mx = fmaxf(mx, __shfl_xor(mx, 8));
      mx = fmaxf(mx, m[r]);
      const float corr = __expf(m[r] - mx);
      const float p0 = __expf(x0 - mx);
      const float p1 = __expf(x1 - mx);
      float sum = p0 + p1;
      sum += __shfl_xor(sum, 1);
      sum += __shfl_xor(sum, 2);
      sum += __shfl_xor(sum, 4);
      sum += __shfl_xor(sum, 8);
      m[r] = mx;
      l[r] = l[r] * corr + sum;
#pragma unroll
      for (int j = 0; j < 4; ++j) O[j][r] *= corr;
      Ps[wave][r + hi * 8][lo]      = (f16)p0;
      Ps[wave][r + hi * 8][16 + lo] = (f16)p1;
    }
    __syncthreads();  // P visible (also all waves done reading Ks this pass)

    // O += P(16x32) x V(32x64): A-frag from Ps, B-frags from Vt.
    const v16h ap = join8(*(const v8h*)&Ps[wave][lo][hi * 8],
                          *(const v8h*)&Ps[wave][lo][16 + hi * 8]);
#pragma unroll
    for (int j = 0; j < 4; ++j) {
      const v16h bv = join8(*(const v8h*)&Vt[16 * j + lo][hi * 8],
                            *(const v8h*)&Vt[16 * j + lo][16 + hi * 8]);
      O[j] = wmma16x16x32(ap, bv, O[j]);
    }
  }

  // Normalize and store f16 to attn buffer laid out as (B, S, H*D) = (B,S,C).
#pragma unroll
  for (int r = 0; r < 8; ++r) {
    const float inv = 1.0f / l[r];
    const int row = qb * 64 + wave * 16 + r + hi * 8;
    const size_t base = ((size_t)b * S_ + row) * C_ + h * D_;
#pragma unroll
    for (int j = 0; j < 4; ++j)
      attn_out[base + j * 16 + lo] = (f16)(O[j][r] * inv);
  }
}

// ---------------------------------------------------------------------------
// Launch
// ---------------------------------------------------------------------------
extern "C" void kernel_launch(void* const* d_in, const int* in_sizes, int n_in,
                              void* d_out, int out_size, void* d_ws, size_t ws_size,
                              hipStream_t stream) {
  const float* x     = (const float*)d_in[0];
  const float* gamma = (const float*)d_in[1];
  const float* beta  = (const float*)d_in[2];
  const float* Wqkv  = (const float*)d_in[3];
  const float* bqkv  = (const float*)d_in[4];
  const float* Wo    = (const float*)d_in[5];
  const float* bo    = (const float*)d_in[6];
  float* out = (float*)d_out;

  char* w = (char*)d_ws;
  size_t off = 0;
  float* stats = (float*)(w + off);  off = 4096;
  f16* hbuf    = (f16*)(w + off);    off += (size_t)B_ * S_ * C_ * 2;        // 16 MB
  f16* wqkv16  = (f16*)(w + off);    off += (size_t)C_ * 3 * C_ * 2;         //  6 MB
  f16* wo16    = (f16*)(w + off);    off += (size_t)C_ * C_ * 2;             //  2 MB
  f16* qkv16   = (f16*)(w + off);    off += (size_t)B_ * S_ * 3 * C_ * 2;    // 48 MB
  f16* attn16  = (f16*)(w + off);    off += (size_t)B_ * S_ * C_ * 2;        // 16 MB

  const int M = B_ * S_;  // 8192

  gn_stats_kernel<<<B_ * NG_, 256, 0, stream>>>(x, stats);
  gn_apply_kernel<<<(M * C_ / 4) / 256, 256, 0, stream>>>(x, gamma, beta, stats, hbuf);
  cvt_f16_kernel<<<(C_ * 3 * C_ / 4 + 255) / 256, 256, 0, stream>>>(Wqkv, wqkv16, C_ * 3 * C_);
  cvt_f16_kernel<<<(C_ * C_ / 4 + 255) / 256, 256, 0, stream>>>(Wo, wo16, C_ * C_);

  gemm_f16_kernel<<<dim3(3 * C_ / 64, M / 128), 128, 0, stream>>>(
      hbuf, wqkv16, bqkv, nullptr, (void*)qkv16, M, 3 * C_, C_, 0);

  attention_kernel<<<dim3(S_ / 64, H_, B_), 128, 0, stream>>>(qkv16, attn16);

  gemm_f16_kernel<<<dim3(C_ / 64, M / 128), 128, 0, stream>>>(
      attn16, wo16, bo, x, (void*)out, M, C_, C_, 1);
}
